// QwenImageAttentionWithRope_11295763988742
// MI455X (gfx1250) — matrix operand order repaired and checked by reading, not compiled, not run on hardware
//
#include <hip/hip_runtime.h>
#include <hip/hip_bf16.h>
#include <math.h>

#define S_TOK   8192
#define BATCH   8
#define LSEQ    1024
#define HID     2048
#define NH      16
#define NKV     4
#define DH      128
#define HALF_D  64
#define QKV_COLS 3072
#define ATT_SCALE 0.08838834764831845f

typedef __attribute__((ext_vector_type(16))) _Float16 v16h;
typedef __attribute__((ext_vector_type(8)))  float    v8f;

#if defined(__has_builtin)
#if __has_builtin(__builtin_amdgcn_global_load_async_to_lds_b128)
#define HAVE_ASYNC_LDS 1   // instruction exists in this toolchain's assembler
#endif
#endif

static __device__ __forceinline__ v8f wmma_f16(v16h a, v16h b, v8f c) {
  // D = A(16x32 f16) * B(32x16 f16) + C(16x16 f32)
  return __builtin_amdgcn_wmma_f32_16x16x32_f16(false, a, false, b, (short)0, c,
                                                false, false);
}

// 16-byte global -> LDS copy; gfx1250 async (ASYNCcnt-tracked) path when the
// toolchain supports it.  LDS operand = low 32 bits of the generic pointer
static __device__ __forceinline__ void copy16_to_lds(const _Float16* g, _Float16* l) {
#ifdef HAVE_ASYNC_LDS
  unsigned lds_addr = (unsigned)(size_t)l;
  unsigned long long gaddr = (unsigned long long)(size_t)g;
  asm volatile("global_load_async_to_lds_b128 %0, %1, off"
               :: "v"(lds_addr), "v"(gaddr) : "memory");
#else
  *(uint4*)l = *(const uint4*)g;
#endif
}

static __device__ __forceinline__ void wait_async_lds() {
#ifdef HAVE_ASYNC_LDS
#if __has_builtin(__builtin_amdgcn_s_wait_asynccnt)
  __builtin_amdgcn_s_wait_asynccnt(0);
#else
  asm volatile("s_wait_asynccnt 0" ::: "memory");
#endif
#endif
}

// ---- fragment loaders ------------------------------------------------------
// A (16x32, f16): lanes 0-15 hold row r=lane, halves {K0..7, K16..23};
//                 lanes 16-31 hold row r=lane-16, halves {K8..15, K24..31}.
static __device__ __forceinline__ v16h frag_a_f16(const _Float16* base, int ld, int k0) {
  int lane = threadIdx.x & 31;
  int r = lane & 15, sel = lane >> 4;
  const _Float16* p = base + (size_t)r * ld + k0 + sel * 8;
  v16h f;
#pragma unroll
  for (int j = 0; j < 8; ++j) f[j] = p[j];
#pragma unroll
  for (int j = 0; j < 8; ++j) f[8 + j] = p[16 + j];
  return f;
}

// B (32x16, f16): lanes 0-15 hold col n=lane, {K0..7, K16..23}; lanes 16-31 col
// n=lane-16, {K8..15, K24..31}.  base points at B^T row-major (memory is [n][k]).
static __device__ __forceinline__ v16h frag_b_f16(const _Float16* base, int ld, int k0) {
  int lane = threadIdx.x & 31;
  int n = lane & 15, sel = lane >> 4;
  const _Float16* p = base + (size_t)n * ld + k0 + sel * 8;
  v16h f;
#pragma unroll
  for (int j = 0; j < 8; ++j) f[j] = p[j];
#pragma unroll
  for (int j = 0; j < 8; ++j) f[8 + j] = p[16 + j];
  return f;
}

// 16-lane (half-wave) row reductions for the WMMA C layout
static __device__ __forceinline__ float rowmax16(float v) {
#pragma unroll
  for (int m = 8; m >= 1; m >>= 1) v = fmaxf(v, __shfl_xor(v, m, 32));
  return v;
}
static __device__ __forceinline__ float rowsum16(float v) {
#pragma unroll
  for (int m = 8; m >= 1; m >>= 1) v += __shfl_xor(v, m, 32);
  return v;
}

// ---- kernel 0: fp32 -> f16 conversion (one pass, removes cvt from GEMM loops)
__global__ void cvt_f16_kernel(const float* __restrict__ src,
                               _Float16* __restrict__ dst, int n) {
  int i = (blockIdx.x * blockDim.x + threadIdx.x) * 4;
  if (i + 3 < n) {
    float4 v = *(const float4*)(src + i);
    dst[i + 0] = (_Float16)v.x;
    dst[i + 1] = (_Float16)v.y;
    dst[i + 2] = (_Float16)v.z;
    dst[i + 3] = (_Float16)v.w;
  }
}

// ---- kernel 1: qkv = x @ [wq;wk;wv]^T + b, LDS-tiled -----------------------
// Block: 256 thr (8 waves) -> 32 rows x 128 cols. A-tile (32x32 f16) staged in
// LDS via async global->LDS, shared by all 8 waves; each wave does two 16x16
// tiles per B-fragment load.
__global__ void qkv_gemm_kernel(const _Float16* __restrict__ xh,
                                const _Float16* __restrict__ wqkv,
                                const float* __restrict__ bq,
                                const float* __restrict__ bk,
                                const float* __restrict__ bv,
                                float* __restrict__ qkv) {
  __shared__ _Float16 asmem[32 * 32];
  int tid = threadIdx.x;
  int wave = tid >> 5;
  int lane = tid & 31;
  int row0 = blockIdx.y * 32;
  int col0 = blockIdx.x * 128 + wave * 16;

  const _Float16* wrow = wqkv + (size_t)col0 * HID;
  v8f acc0 = {}, acc1 = {};
  for (int kk = 0; kk < HID; kk += 32) {
    __syncthreads();
    if (tid < 128) {
      int r = tid >> 2, c = (tid & 3) * 8;
      copy16_to_lds(xh + (size_t)(row0 + r) * HID + kk + c, &asmem[r * 32 + c]);
    }
    wait_async_lds();
    __syncthreads();
    __builtin_prefetch(wrow + kk + 256, 0, 0);
    v16h a0 = frag_a_f16(asmem, 32, 0);
    v16h a1 = frag_a_f16(asmem + 16 * 32, 32, 0);
    v16h b  = frag_b_f16(wrow, HID, kk);
    acc0 = wmma_f16(a0, b, acc0);
    acc1 = wmma_f16(a1, b, acc1);
  }

  const float* brow;
  if (col0 < NH * DH)                { brow = bq + col0; }
  else if (col0 < NH*DH + NKV*DH)    { brow = bk + (col0 - NH*DH); }
  else                               { brow = bv + (col0 - NH*DH - NKV*DH); }
  int n = lane & 15, sel = lane >> 4;
  float bias = brow[n];
#pragma unroll
  for (int i = 0; i < 8; ++i) {
    int m = i + sel * 8;
    qkv[(size_t)(row0 + m) * QKV_COLS + col0 + n]      = acc0[i] + bias;
    qkv[(size_t)(row0 + 16 + m) * QKV_COLS + col0 + n] = acc1[i] + bias;
  }
}

// ---- kernel 2/3: M-RoPE on q / k, pack to f16 head-major layouts ----------
__global__ void rope_q_kernel(const float* __restrict__ qkv,
                              const float* __restrict__ freqs,
                              const int* __restrict__ pos_ids,
                              _Float16* __restrict__ qf) {
  int s = blockIdx.x;
  int h = threadIdx.x >> 6;
  int d = threadIdx.x & 63;
  float q1 = qkv[(size_t)s * QKV_COLS + h * DH + d];
  float q2 = qkv[(size_t)s * QKV_COLS + h * DH + d + HALF_D];
  int row = (d < 16) ? 0 : ((d < 40) ? 1 : 2);   // mrope sections 16/24/24
  int pos = pos_ids[row * S_TOK + s];
  float c  = freqs[((size_t)pos * HALF_D + d) * 2];
  float si = freqs[((size_t)pos * HALF_D + d) * 2 + 1];
  float o1 = q1 * c - q2 * si;
  float o2 = q2 * c + q1 * si;
  int b = s >> 10, l = s & 1023;
  size_t base = (((size_t)b * NH + h) * LSEQ + l) * DH;
  qf[base + d]          = (_Float16)o1;
  qf[base + d + HALF_D] = (_Float16)o2;
}

__global__ void rope_k_kernel(const float* __restrict__ qkv,
                              const float* __restrict__ freqs,
                              const int* __restrict__ pos_ids,
                              _Float16* __restrict__ kf) {
  int s = blockIdx.x;
  int kv = threadIdx.x >> 6;
  int d  = threadIdx.x & 63;
  float k1 = qkv[(size_t)s * QKV_COLS + NH * DH + kv * DH + d];
  float k2 = qkv[(size_t)s * QKV_COLS + NH * DH + kv * DH + d + HALF_D];
  int row = (d < 16) ? 0 : ((d < 40) ? 1 : 2);
  int pos = pos_ids[row * S_TOK + s];
  float c  = freqs[((size_t)pos * HALF_D + d) * 2];
  float si = freqs[((size_t)pos * HALF_D + d) * 2 + 1];
  float o1 = k1 * c - k2 * si;
  float o2 = k2 * c + k1 * si;
  int b = s >> 10, l = s & 1023;
  size_t base = (((size_t)b * NKV + kv) * LSEQ + l) * DH;
  kf[base + d]          = (_Float16)o1;
  kf[base + d + HALF_D] = (_Float16)o2;
}

// ---- kernel 4: pack V transposed [B, KV, D, L] f16 -------------------------
__global__ void pack_v_kernel(const float* __restrict__ qkv,
                              _Float16* __restrict__ vT) {
  int s = blockIdx.x;
  int b = s >> 10, l = s & 1023;
  for (int t = threadIdx.x; t < NKV * DH; t += blockDim.x) {
    int kv = t >> 7, d = t & 127;
    float v = qkv[(size_t)s * QKV_COLS + NH * DH + NKV * DH + t];
    vT[(((size_t)b * NKV + kv) * DH + d) * LSEQ + l] = (_Float16)v;
  }
}

// ---- kernel 5: causal GQA flash attention ---------------------------------
// 4 waves/block; each wave owns one (b,h,16-query tile) and a private LDS
// P-slice.  No block barriers (divergent causal trip counts); per-wave DS
// ordering + wave_barrier scheduling fences guarantee the LDS P staging.
__global__ void attention_kernel(const _Float16* __restrict__ qf,
                                 const _Float16* __restrict__ kf,
                                 const _Float16* __restrict__ vT,
                                 _Float16* __restrict__ attn) {
  __shared__ _Float16 pbuf[4][16 * 32];
  int bid = blockIdx.x;
  int wave = threadIdx.x >> 5;
  int qt = (bid & 15) * 4 + wave;
  int h  = (bid >> 4) & 15;
  int b  = bid >> 8;
  int kvh = h >> 2;                       // GQA: 4 q-heads per kv-head
  int lane = threadIdx.x & 31;
  int n = lane & 15, sel = lane >> 4;
  int qrow0 = qt * 16;
  _Float16* pw = &pbuf[wave][0];

  const _Float16* qhead = qf + (((size_t)b * NH + h) * LSEQ + qrow0) * DH;
  const _Float16* khead = kf + (((size_t)b * NKV + kvh) * LSEQ) * DH;
  const _Float16* vhead = vT + (((size_t)b * NKV + kvh) * DH) * LSEQ;

  v16h qa[4];
#pragma unroll
  for (int t = 0; t < 4; ++t) qa[t] = frag_a_f16(qhead, DH, t * 32);

  v8f O[8];
#pragma unroll
  for (int t = 0; t < 8; ++t) O[t] = (v8f){};
  float rowm[8], rowl[8];
#pragma unroll
  for (int i = 0; i < 8; ++i) { rowm[i] = -1e30f; rowl[i] = 0.f; }

  int nch = (qrow0 + 16 + 31) >> 5;       // 32-key chunks under causality
  for (int ch = 0; ch < nch; ++ch) {
    int key0 = ch * 32;
    v8f s0 = (v8f){}, s1 = (v8f){};
#pragma unroll
    for (int t = 0; t < 4; ++t) {
      v16h kb0 = frag_b_f16(khead + (size_t)key0 * DH, DH, t * 32);
      v16h kb1 = frag_b_f16(khead + (size_t)(key0 + 16) * DH, DH, t * 32);
      s0 = wmma_f16(qa[t], kb0, s0);
      s1 = wmma_f16(qa[t], kb1, s1);
    }
    // online softmax (rows live across 16-lane halves in C layout)
#pragma unroll
    for (int i = 0; i < 8; ++i) {
      int q = qrow0 + i + sel * 8;
      float v0 = s0[i] * ATT_SCALE;
      float v1 = s1[i] * ATT_SCALE;
      if (key0 + n > q)      v0 = -1e30f;
      if (key0 + 16 + n > q) v1 = -1e30f;
      float rm = rowmax16(fmaxf(v0, v1));
      rm = fmaxf(rm, rowm[i]);
      float alpha = __expf(rowm[i] - rm);
      rowm[i] = rm;
      float p0 = __expf(v0 - rm);
      float p1 = __expf(v1 - rm);
      rowl[i] = rowl[i] * alpha + rowsum16(p0 + p1);
#pragma unroll
      for (int t = 0; t < 8; ++t) O[t][i] *= alpha;
      int r = i + sel * 8;
      pw[r * 32 + n]      = (_Float16)p0;
      pw[r * 32 + 16 + n] = (_Float16)p1;
    }
    __builtin_amdgcn_wave_barrier();      // keep ds stores before ds loads
    v16h pa = frag_a_f16(pw, 32, 0);      // C-layout -> A-fragment via LDS
#pragma unroll
    for (int t = 0; t < 8; ++t) {
      v16h vb = frag_b_f16(vhead + (size_t)(t * 16) * LSEQ, LSEQ, key0);
      O[t] = wmma_f16(pa, vb, O[t]);
    }
    __builtin_amdgcn_wave_barrier();
  }

#pragma unroll
  for (int i = 0; i < 8; ++i) {
    float inv = 1.0f / rowl[i];
    int m = i + sel * 8;
    size_t srow = (size_t)(b * LSEQ + qrow0 + m) * (NH * DH) + h * DH;
#pragma unroll
    for (int t = 0; t < 8; ++t)
      attn[srow + t * 16 + n] = (_Float16)(O[t][i] * inv);
  }
}

// ---- kernel 6: out = attn @ wo^T, LDS-tiled --------------------------------
__global__ void out_gemm_kernel(const _Float16* __restrict__ attn,
                                const _Float16* __restrict__ woh,
                                float* __restrict__ out) {
  __shared__ _Float16 asmem[32 * 32];
  int tid = threadIdx.x;
  int wave = tid >> 5;
  int lane = tid & 31;
  int row0 = blockIdx.y * 32;
  int col0 = blockIdx.x * 128 + wave * 16;
  const _Float16* wrow = woh + (size_t)col0 * HID;

  v8f acc0 = {}, acc1 = {};
  for (int kk = 0; kk < HID; kk += 32) {
    __syncthreads();
    if (tid < 128) {
      int r = tid >> 2, c = (tid & 3) * 8;
      copy16_to_lds(attn + (size_t)(row0 + r) * HID + kk + c, &asmem[r * 32 + c]);
    }
    wait_async_lds();
    __syncthreads();
    __builtin_prefetch(wrow + kk + 256, 0, 0);
    v16h a0 = frag_a_f16(asmem, 32, 0);
    v16h a1 = frag_a_f16(asmem + 16 * 32, 32, 0);
    v16h b  = frag_b_f16(wrow, HID, kk);
    acc0 = wmma_f16(a0, b, acc0);
    acc1 = wmma_f16(a1, b, acc1);
  }
  int n = lane & 15, sel = lane >> 4;
#pragma unroll
  for (int i = 0; i < 8; ++i) {
    int m = i + sel * 8;
    out[(size_t)(row0 + m) * HID + col0 + n]      = acc0[i];
    out[(size_t)(row0 + 16 + m) * HID + col0 + n] = acc1[i];
  }
}

// ---- host-side orchestration ----------------------------------------------
extern "C" void kernel_launch(void* const* d_in, const int* in_sizes, int n_in,
                              void* d_out, int out_size, void* d_ws, size_t ws_size,
                              hipStream_t stream) {
  const float* x     = (const float*)d_in[0];
  const float* wq    = (const float*)d_in[1];
  const float* wk    = (const float*)d_in[2];
  const float* wv    = (const float*)d_in[3];
  const float* bq    = (const float*)d_in[4];
  const float* bk    = (const float*)d_in[5];
  const float* bv    = (const float*)d_in[6];
  const float* wo    = (const float*)d_in[7];
  const float* freqs = (const float*)d_in[8];
  const int* pos_ids = (const int*)d_in[9];
  // d_in[10] input_row_offsets: uniform L, not needed
  float* out = (float*)d_out;

  char* w = (char*)d_ws;
  _Float16* xh    = (_Float16*)w; w += (size_t)S_TOK * HID      * sizeof(_Float16); // 32 MB
  _Float16* wqkvh = (_Float16*)w; w += (size_t)QKV_COLS * HID   * sizeof(_Float16); // 12 MB
  _Float16* woh   = (_Float16*)w; w += (size_t)HID * HID        * sizeof(_Float16); //  8 MB
  float*    qkv   = (float*)w;    w += (size_t)S_TOK * QKV_COLS * sizeof(float);    // 96 MB
  _Float16* qf    = (_Float16*)w; w += (size_t)S_TOK * NH  * DH * sizeof(_Float16); // 32 MB
  _Float16* kf    = (_Float16*)w; w += (size_t)S_TOK * NKV * DH * sizeof(_Float16); //  8 MB
  _Float16* vT    = (_Float16*)w; w += (size_t)S_TOK * NKV * DH * sizeof(_Float16); //  8 MB
  _Float16* attn  = (_Float16*)w; w += (size_t)S_TOK * NH  * DH * sizeof(_Float16); // 32 MB

  // one-time fp32 -> f16 packs
  cvt_f16_kernel<<<(S_TOK * HID) / 1024, 256, 0, stream>>>(x, xh, S_TOK * HID);
  cvt_f16_kernel<<<(NH * DH * HID) / 1024, 256, 0, stream>>>(wq, wqkvh, NH * DH * HID);
  cvt_f16_kernel<<<(NKV * DH * HID) / 1024, 256, 0, stream>>>(
      wk, wqkvh + (size_t)NH * DH * HID, NKV * DH * HID);
  cvt_f16_kernel<<<(NKV * DH * HID) / 1024, 256, 0, stream>>>(
      wv, wqkvh + (size_t)(NH + NKV) * DH * HID, NKV * DH * HID);
  cvt_f16_kernel<<<(HID * HID) / 1024, 256, 0, stream>>>(wo, woh, HID * HID);

  qkv_gemm_kernel<<<dim3(QKV_COLS / 128, S_TOK / 32), 256, 0, stream>>>(
      xh, wqkvh, bq, bk, bv, qkv);
  rope_q_kernel<<<S_TOK, NH * HALF_D, 0, stream>>>(qkv, freqs, pos_ids, qf);
  rope_k_kernel<<<S_TOK, NKV * HALF_D, 0, stream>>>(qkv, freqs, pos_ids, kf);
  pack_v_kernel<<<S_TOK, 256, 0, stream>>>(qkv, vT);
  attention_kernel<<<BATCH * NH * (LSEQ / 64), 128, 0, stream>>>(qf, kf, vT, attn);
  out_gemm_kernel<<<dim3(HID / 128, S_TOK / 32), 256, 0, stream>>>(attn, woh, out);
}
